// Dilated2DBEVBackboneBlockSequence_25443386261908
// MI455X (gfx1250) — matrix operand order repaired and checked by reading, not compiled, not run on hardware
//
#include <hip/hip_runtime.h>

// ---------------------------------------------------------------------------
// Swin block sequence (D=2) for MI455X / gfx1250, wave32 + WMMA bf16.
//   WS=7, N=49 (pad to 64), C=192, H=6, HD=32, F=768, BW=8192
// All GEMMs via v_wmma_f32_16x16x32_bf16. Weights pre-converted to bf16 once
// (L2-resident), output-tile pairing for independent WMMA chains.
// ---------------------------------------------------------------------------

#define BW_TOK 49
#define NPAD   64
#define CDIM   192
#define NHEAD  6
#define HDIM   32
#define FDIM   768

typedef __attribute__((ext_vector_type(16))) __bf16 v16bf;
typedef __attribute__((ext_vector_type(8)))  __bf16 v8bf;
typedef __attribute__((ext_vector_type(4)))  __bf16 v4bf;
typedef __attribute__((ext_vector_type(8)))  float  v8f;
typedef __attribute__((ext_vector_type(4)))  float  f32x4;

__device__ __forceinline__ v8f zero8() {
  v8f z;
#pragma unroll
  for (int i = 0; i < 8; ++i) z[i] = 0.f;
  return z;
}

__device__ __forceinline__ v8f wmma_bf16(v16bf a, v16bf b, v8f c) {
  return __builtin_amdgcn_wmma_f32_16x16x32_bf16(
      /*neg_a=*/false, a, /*neg_b=*/false, b,
      /*c_mod=*/(short)0, c, /*reuse_a=*/false, /*reuse_b=*/false);
}

// A-matrix fragment (16x32 bf16) from LDS, row-major [M][K], stride in elems.
// Layout: lane(r,hf): f[0..15]=K(hf*16 + 0..15).
__device__ __forceinline__ v16bf fragA_lds(const __bf16* base, int stride) {
  const int lane = threadIdx.x & 31;
  const int hf = lane >> 4;
  const int r  = lane & 15;
  const __bf16* p = base + r * stride + hf * 16;
  v8bf lo = *(const v8bf*)(p);
  v8bf hi = *(const v8bf*)(p + 8);
  v16bf f;
#pragma unroll
  for (int i = 0; i < 8; ++i) { f[i] = lo[i]; f[i + 8] = hi[i]; }
  return f;
}

// B-matrix fragment (32x16 bf16) from LDS holding W row-major [Nout][K].
// Contiguous k order, same as fragA: lanes 0-15 hold K=0..15, lanes 16-31 hold K=16..31, lane=N.
__device__ __forceinline__ v16bf fragB_lds(const __bf16* base, int stride) {
  const int lane = threadIdx.x & 31;
  const int hf = lane >> 4;
  const int n  = lane & 15;
  const __bf16* p = base + n * stride + hf * 16;
  v8bf lo = *(const v8bf*)(p);
  v8bf hi = *(const v8bf*)(p + 8);
  v16bf f;
#pragma unroll
  for (int i = 0; i < 8; ++i) { f[i] = lo[i]; f[i + 8] = hi[i]; }
  return f;
}

// B-matrix fragment from pre-converted global bf16 weight [Nout][K].
__device__ __forceinline__ v16bf fragB_gbf(const __bf16* base, int stride) {
  const int lane = threadIdx.x & 31;
  const int hf = lane >> 4;
  const int n  = lane & 15;
  const __bf16* p = base + n * stride + hf * 16;
  v8bf lo = *(const v8bf*)(p);
  v8bf hi = *(const v8bf*)(p + 8);
  v16bf f;
#pragma unroll
  for (int i = 0; i < 8; ++i) { f[i] = lo[i]; f[i + 8] = hi[i]; }
  return f;
}

// One-time f32 -> bf16 weight conversion (4 elems/thread, count % 1024 == 0).
__global__ void __launch_bounds__(256)
cvt_f32_to_bf16(const float* __restrict__ s, __bf16* __restrict__ d) {
  const int i = (blockIdx.x * 256 + threadIdx.x) * 4;
  f32x4 v = *(const f32x4*)(s + i);
  v4bf o;
#pragma unroll
  for (int j = 0; j < 4; ++j) o[j] = (__bf16)v[j];
  *(v4bf*)(d + i) = o;
}

// LayerNorm of 49x192 -> bf16 LDS [64][192], pad rows zeroed.
// 4 threads per row, 64 rows, 256 threads.
__device__ __forceinline__ void layernorm_to_lds(
    const float* __restrict__ xg, const float* __restrict__ gw,
    const float* __restrict__ gb, __bf16* hN) {
  const int t = threadIdx.x;
  const int row = t >> 2;
  const int sub = t & 3;
  float s = 0.f, ss = 0.f;
  const float* p = xg + row * CDIM + sub * 48;
  if (row < BW_TOK) {
#pragma unroll 8
    for (int j = 0; j < 48; ++j) { float v = p[j]; s += v; ss += v * v; }
  }
  s += __shfl_xor(s, 1, 32);  ss += __shfl_xor(ss, 1, 32);
  s += __shfl_xor(s, 2, 32);  ss += __shfl_xor(ss, 2, 32);
  const float mean = s * (1.f / CDIM);
  const float var  = ss * (1.f / CDIM) - mean * mean;
  const float rstd = rsqrtf(var + 1e-5f);
  __bf16* o = hN + row * CDIM + sub * 48;
  if (row < BW_TOK) {
#pragma unroll 8
    for (int j = 0; j < 48; ++j) {
      float v = (p[j] - mean) * rstd * gw[sub * 48 + j] + gb[sub * 48 + j];
      o[j] = (__bf16)v;
    }
  } else {
#pragma unroll 8
    for (int j = 0; j < 48; ++j) o[j] = (__bf16)0.f;
  }
}

// ---------------------------------------------------------------------------
// Kernel A: LN1 + QKV + windowed attention + proj + residual. One window/WG.
// Dynamic LDS layout (192 KB):
//   [0      , 24576) hN  bf16[64][192]   -> reused as O bf16[64][192]
//   [24576  , 49152) q   bf16[64][192]   -> P head h at byte 24576+h*8192
//   [49152  , 73728) k   bf16[64][192]      (P spans q+k regions, 48 KB)
//   [73728  , 98304) vT  bf16[192][64]
//   [98304  ,196608) S   f32 [6][64][64]
// ---------------------------------------------------------------------------
#define SMEM_ATTN 196608

__device__ __forceinline__ void qkv_epilogue(v8f acc, int mt, int nt, int n, int mb,
                                             const float* __restrict__ qkvb,
                                             __bf16* qs, __bf16* ks, __bf16* vT) {
  const int g = nt * 16 + n;
  const float bias = qkvb[g];
#pragma unroll
  for (int r = 0; r < 8; ++r) {
    const int m = mt * 16 + mb + r;
    float v = acc[r] + bias;
    if (g < CDIM)            qs[m * CDIM + g] = (__bf16)(v * 0.17677669529f);
    else if (g < 2 * CDIM)   ks[m * CDIM + (g - CDIM)] = (__bf16)v;
    else                     vT[(g - 2 * CDIM) * NPAD + m] = (__bf16)v;  // V^T
  }
}

__global__ void __launch_bounds__(256)
swin_attn_kernel(const float* __restrict__ x_in,
                 const float* __restrict__ n1w, const float* __restrict__ n1b,
                 const __bf16* __restrict__ qkvw, const float* __restrict__ qkvb,
                 const float* __restrict__ relb,
                 const __bf16* __restrict__ pw, const float* __restrict__ pb,
                 float* __restrict__ x_out) {
  extern __shared__ char smem[];
  __bf16* hN = (__bf16*)smem;                 // later: O
  __bf16* qs = (__bf16*)(smem + 24576);       // later: P
  __bf16* ks = (__bf16*)(smem + 49152);
  __bf16* vT = (__bf16*)(smem + 73728);
  float*  Sb = (float*)(smem + 98304);

  const size_t boff = (size_t)blockIdx.x * (BW_TOK * CDIM);
  const int w    = threadIdx.x >> 5;
  const int lane = threadIdx.x & 31;
  const int n    = lane & 15;
  const int mb   = (lane >> 4) * 8;

  // ---- LN1 ----
  layernorm_to_lds(x_in + boff, n1w, n1b, hN);
  __syncthreads();

  // ---- QKV: [64x192] @ [192x576]^T; 72 tile-pairs sharing A over 8 waves ----
  for (int pi = w; pi < 72; pi += 8) {
    const int mt  = pi / 18;
    const int nt0 = (pi % 18) * 2;
    v8f acc0 = zero8(), acc1 = zero8();
#pragma unroll
    for (int k0 = 0; k0 < 6; ++k0) {
      v16bf A  = fragA_lds(hN + mt * 16 * CDIM + k0 * 32, CDIM);
      v16bf B0 = fragB_gbf(qkvw + (nt0 * 16) * CDIM + k0 * 32, CDIM);
      v16bf B1 = fragB_gbf(qkvw + ((nt0 + 1) * 16) * CDIM + k0 * 32, CDIM);
      acc0 = wmma_bf16(A, B0, acc0);
      acc1 = wmma_bf16(A, B1, acc1);
    }
    qkv_epilogue(acc0, mt, nt0, n, mb, qkvb, qs, ks, vT);
    qkv_epilogue(acc1, mt, nt0 + 1, n, mb, qkvb, qs, ks, vT);
  }
  __syncthreads();

  // ---- S = q k^T + rel_bias (wave w = head w); K-fragments register-cached ----
  if (w < NHEAD) {
    const int h = w;
    float* S = Sb + h * (NPAD * NPAD);
    v16bf Bk[4];
#pragma unroll
    for (int nt = 0; nt < 4; ++nt)
      Bk[nt] = fragB_lds(ks + nt * 16 * CDIM + h * HDIM, CDIM);
#pragma unroll
    for (int mt = 0; mt < 4; ++mt) {
      v16bf A = fragA_lds(qs + mt * 16 * CDIM + h * HDIM, CDIM);
#pragma unroll
      for (int nt = 0; nt < 4; ++nt) {
        v8f acc = wmma_bf16(A, Bk[nt], zero8());
        const int kt = nt * 16 + n;
#pragma unroll
        for (int r = 0; r < 8; ++r) {
          const int qt = mt * 16 + mb + r;
          float v = acc[r];
          if (kt >= BW_TOK) v = -1e30f;
          else if (qt < BW_TOK) {
            const int qy = qt / 7, qx = qt - qy * 7;
            const int ky = kt / 7, kx = kt - ky * 7;
            const int idx = (qy - ky + 6) * 13 + (qx - kx + 6);
            v += relb[idx * NHEAD + h];
          }
          S[qt * NPAD + kt] = v;
        }
      }
    }
  }
  __syncthreads();

  // ---- softmax rows (P bf16 overlays dead q/k regions) ----
  if (w < NHEAD) {
    const int h = w;
    const float* S = Sb + h * (NPAD * NPAD);
    __bf16* P = qs + h * (NPAD * NPAD);
#pragma unroll
    for (int it = 0; it < 2; ++it) {
      const int row = it * 32 + lane;
      float buf[NPAD];
      float mx = -1e30f;
#pragma unroll
      for (int j = 0; j < NPAD; ++j) { buf[j] = S[row * NPAD + j]; mx = fmaxf(mx, buf[j]); }
      float sum = 0.f;
#pragma unroll
      for (int j = 0; j < NPAD; ++j) { float e = __expf(buf[j] - mx); buf[j] = e; sum += e; }
      const float inv = 1.f / sum;
#pragma unroll
      for (int j = 0; j < NPAD; ++j) P[row * NPAD + j] = (__bf16)(buf[j] * inv);
    }
  }
  __syncthreads();

  // ---- O = P V (O overlays dead hN); V-fragments register-cached ----
  __bf16* Ob = hN;
  if (w < NHEAD) {
    const int h = w;
    const __bf16* P = qs + h * (NPAD * NPAD);
    v16bf Bv[4];  // [nt][k0]
#pragma unroll
    for (int nt = 0; nt < 2; ++nt)
#pragma unroll
      for (int k0 = 0; k0 < 2; ++k0)
        Bv[nt * 2 + k0] = fragB_lds(vT + (h * HDIM + nt * 16) * NPAD + k0 * 32, NPAD);
#pragma unroll
    for (int mt = 0; mt < 4; ++mt) {
      v16bf A0 = fragA_lds(P + mt * 16 * NPAD, NPAD);
      v16bf A1 = fragA_lds(P + mt * 16 * NPAD + 32, NPAD);
#pragma unroll
      for (int nt = 0; nt < 2; ++nt) {
        v8f acc = wmma_bf16(A0, Bv[nt * 2], zero8());
        acc = wmma_bf16(A1, Bv[nt * 2 + 1], acc);
#pragma unroll
        for (int r = 0; r < 8; ++r)
          Ob[(mt * 16 + mb + r) * CDIM + h * HDIM + nt * 16 + n] = (__bf16)acc[r];
      }
    }
  }
  __syncthreads();

  // ---- proj + residual: [64x192] @ [192x192]^T; 24 tile-pairs / 8 waves ----
  for (int pi = w; pi < 24; pi += 8) {
    const int mt  = pi / 6;
    const int nt0 = (pi % 6) * 2;
    v8f acc0 = zero8(), acc1 = zero8();
#pragma unroll
    for (int k0 = 0; k0 < 6; ++k0) {
      v16bf A  = fragA_lds(Ob + mt * 16 * CDIM + k0 * 32, CDIM);
      v16bf B0 = fragB_gbf(pw + (nt0 * 16) * CDIM + k0 * 32, CDIM);
      v16bf B1 = fragB_gbf(pw + ((nt0 + 1) * 16) * CDIM + k0 * 32, CDIM);
      acc0 = wmma_bf16(A, B0, acc0);
      acc1 = wmma_bf16(A, B1, acc1);
    }
#pragma unroll
    for (int q = 0; q < 2; ++q) {
      const v8f acc = q ? acc1 : acc0;
      const int col = (nt0 + q) * 16 + n;
      const float bv = pb[col];
#pragma unroll
      for (int r = 0; r < 8; ++r) {
        const int row = mt * 16 + mb + r;
        if (row < BW_TOK)
          x_out[boff + row * CDIM + col] = acc[r] + bv + x_in[boff + row * CDIM + col];
      }
    }
  }
}

// ---------------------------------------------------------------------------
// Kernel B: LN2 + FFN (192->768 GELU 768->192) + residual, chunked over F.
// LDS (32 KB): hN bf16[64][192] @0, G bf16[64][64] @24576.
// O-tile accumulators (3 pairs/wave) live in registers across all F-chunks.
// ---------------------------------------------------------------------------
#define SMEM_FFN 32768

__global__ void __launch_bounds__(256)
swin_ffn_kernel(const float* __restrict__ x_in,
                const float* __restrict__ n2w, const float* __restrict__ n2b,
                const __bf16* __restrict__ w1, const float* __restrict__ b1,
                const __bf16* __restrict__ w2, const float* __restrict__ b2,
                float* __restrict__ x_out) {
  extern __shared__ char smem[];
  __bf16* hN = (__bf16*)smem;
  __bf16* G  = (__bf16*)(smem + 24576);

  const size_t boff = (size_t)blockIdx.x * (BW_TOK * CDIM);
  const int w    = threadIdx.x >> 5;
  const int lane = threadIdx.x & 31;
  const int n    = lane & 15;
  const int mb   = (lane >> 4) * 8;

  layernorm_to_lds(x_in + boff, n2w, n2b, hN);
  __syncthreads();

  v8f accO[6];
#pragma unroll
  for (int j = 0; j < 6; ++j) accO[j] = zero8();

  for (int fc = 0; fc < FDIM / 64; ++fc) {
    // GEMM1 chunk + GELU -> G bf16 [64][64]; one tile-pair per wave
    {
      const int gm  = w >> 1;
      const int gn0 = (w & 1) * 2;
      v8f a0 = zero8(), a1 = zero8();
#pragma unroll
      for (int k0 = 0; k0 < 6; ++k0) {
        v16bf A  = fragA_lds(hN + gm * 16 * CDIM + k0 * 32, CDIM);
        v16bf B0 = fragB_gbf(w1 + (fc * 64 + gn0 * 16) * CDIM + k0 * 32, CDIM);
        v16bf B1 = fragB_gbf(w1 + (fc * 64 + (gn0 + 1) * 16) * CDIM + k0 * 32, CDIM);
        a0 = wmma_bf16(A, B0, a0);
        a1 = wmma_bf16(A, B1, a1);
      }
#pragma unroll
      for (int q = 0; q < 2; ++q) {
        const v8f acc = q ? a1 : a0;
        const int gn = gn0 + q;
        const float bb = b1[fc * 64 + gn * 16 + n];
#pragma unroll
        for (int r = 0; r < 8; ++r) {
          float v = acc[r] + bb;
          v = 0.5f * v * (1.f + erff(v * 0.70710678118f));  // exact GELU
          G[(gm * 16 + mb + r) * 64 + gn * 16 + n] = (__bf16)v;
        }
      }
    }
    __syncthreads();
    // GEMM2 accumulate: O_tiles += G_chunk @ W2(:, fc-slice)^T; 3 pairs/wave
#pragma unroll
    for (int j = 0; j < 3; ++j) {
      const int pi  = w + 8 * j;
      const int mt  = pi / 6;
      const int nt0 = (pi % 6) * 2;
#pragma unroll
      for (int k0 = 0; k0 < 2; ++k0) {
        v16bf A  = fragA_lds(G + mt * 16 * 64 + k0 * 32, 64);
        v16bf B0 = fragB_gbf(w2 + (nt0 * 16) * FDIM + fc * 64 + k0 * 32, FDIM);
        v16bf B1 = fragB_gbf(w2 + ((nt0 + 1) * 16) * FDIM + fc * 64 + k0 * 32, FDIM);
        accO[2 * j]     = wmma_bf16(A, B0, accO[2 * j]);
        accO[2 * j + 1] = wmma_bf16(A, B1, accO[2 * j + 1]);
      }
    }
    __syncthreads();
  }

  // epilogue: bias + residual
#pragma unroll
  for (int j = 0; j < 3; ++j) {
    const int pi  = w + 8 * j;
    const int mt  = pi / 6;
    const int nt0 = (pi % 6) * 2;
#pragma unroll
    for (int q = 0; q < 2; ++q) {
      const int col = (nt0 + q) * 16 + n;
      const float bb = b2[col];
#pragma unroll
      for (int r = 0; r < 8; ++r) {
        const int row = mt * 16 + mb + r;
        if (row < BW_TOK)
          x_out[boff + row * CDIM + col] =
              accO[2 * j + q][r] + bb + x_in[boff + row * CDIM + col];
      }
    }
  }
}

// ---------------------------------------------------------------------------
extern "C" void kernel_launch(void* const* d_in, const int* in_sizes, int n_in,
                              void* d_out, int out_size, void* d_ws, size_t ws_size,
                              hipStream_t stream) {
  const float* x   = (const float*)d_in[0];
  // d_in[1] = window_indices (unused: heightmerging is None)
  const float* n1w = (const float*)d_in[2];
  const float* n1b = (const float*)d_in[3];
  const float* qw  = (const float*)d_in[4];
  const float* qb  = (const float*)d_in[5];
  const float* rb  = (const float*)d_in[6];
  const float* pw  = (const float*)d_in[7];
  const float* pb  = (const float*)d_in[8];
  const float* n2w = (const float*)d_in[9];
  const float* n2b = (const float*)d_in[10];
  const float* w1  = (const float*)d_in[11];
  const float* b1  = (const float*)d_in[12];
  const float* w2  = (const float*)d_in[13];
  const float* b2  = (const float*)d_in[14];
  float* out = (float*)d_out;

  const int BW = in_sizes[0] / (BW_TOK * CDIM);

  // workspace layout: [ x ping buffer f32 | bf16 weights ]
  float* xs = (float*)d_ws;
  __bf16* wb = (__bf16*)((char*)d_ws + (size_t)BW * BW_TOK * CDIM * sizeof(float));
  __bf16* qwb = wb;                         // 2*576*192 = 221184
  __bf16* pwb = wb + 221184;                // 2*192*192 =  73728
  __bf16* w1b = wb + 294912;                // 2*768*192 = 294912
  __bf16* w2b = wb + 589824;                // 2*192*768 = 294912

  (void)hipFuncSetAttribute((const void*)swin_attn_kernel,
                            hipFuncAttributeMaxDynamicSharedMemorySize, SMEM_ATTN);
  (void)hipFuncSetAttribute((const void*)swin_ffn_kernel,
                            hipFuncAttributeMaxDynamicSharedMemorySize, SMEM_FFN);

  // one-time weight conversion f32 -> bf16 (counts are multiples of 1024)
  cvt_f32_to_bf16<<<dim3(216), dim3(256), 0, stream>>>(qw, qwb);
  cvt_f32_to_bf16<<<dim3(72),  dim3(256), 0, stream>>>(pw, pwb);
  cvt_f32_to_bf16<<<dim3(288), dim3(256), 0, stream>>>(w1, w1b);
  cvt_f32_to_bf16<<<dim3(288), dim3(256), 0, stream>>>(w2, w2b);

  dim3 grid(BW), block(256);
  for (int layer = 0; layer < 2; ++layer) {
    const float* xin = (layer == 0) ? x : out;
    swin_attn_kernel<<<grid, block, SMEM_ATTN, stream>>>(
        xin,
        n1w + layer * CDIM, n1b + layer * CDIM,
        qwb + layer * 3 * CDIM * CDIM, qb + layer * 3 * CDIM,
        rb + layer * 169 * NHEAD,
        pwb + layer * CDIM * CDIM, pb + layer * CDIM,
        xs);
    swin_ffn_kernel<<<grid, block, SMEM_FFN, stream>>>(
        xs,
        n2w + layer * CDIM, n2b + layer * CDIM,
        w1b + layer * FDIM * CDIM, b1 + layer * FDIM,
        w2b + layer * CDIM * FDIM, b2 + layer * CDIM,
        out);
  }
}
